// ReprogrammingLayer_46729244180526
// MI455X (gfx1250) — matrix operand order repaired and checked by reading, not compiled, not run on hardware
//
#include <hip/hip_runtime.h>
#include <hip/hip_bf16.h>
#include <math.h>

// ---------------------------------------------------------------------------
// Types for CDNA5 WMMA (gfx1250, wave32)
// ---------------------------------------------------------------------------
typedef __bf16 bf16_t;
typedef __attribute__((ext_vector_type(16))) __bf16 v16bf;
typedef __attribute__((ext_vector_type(8)))  __bf16 v8bf;
typedef __attribute__((ext_vector_type(8)))  float  v8f;

template <typename T>
struct __attribute__((aligned(sizeof(T) * 4))) V4 { T v[4]; };
struct __attribute__((aligned(8))) bf16x4 { bf16_t v[4]; };

static __device__ inline v16bf cat16(v8bf a, v8bf b) {
  return __builtin_shufflevector(a, b, 0,1,2,3,4,5,6,7,8,9,10,11,12,13,14,15);
}

static __device__ inline v8f wmma_bf16(v16bf a, v16bf b, v8f c) {
  // D = A(16x32) * B(32x16) + C, f32 accumulate
  return __builtin_amdgcn_wmma_f32_16x16x32_bf16(
      /*neg_a=*/false, a, /*neg_b=*/false, b,
      /*c_mod=*/(short)0, c, /*reuse_a=*/false, /*reuse_b=*/false);
}

static __device__ inline bf16_t cvt_bf16(float x)  { return (bf16_t)x; }
static __device__ inline bf16_t cvt_bf16(bf16_t x) { return x; }

static __device__ inline void store_out(float* p,  float v) { *p = v; }
static __device__ inline void store_out(bf16_t* p, float v) { *p = (bf16_t)v; }

// ---------------------------------------------------------------------------
// Tiled GEMM:  C[M,N] = cast_bf16(A[M,K]) @ cast_bf16(W[K,N]) + bias[N]
// Block = 256 threads (8 waves), tile 128x128, K-step 32, double-buffered LDS.
// Wave w: rows 32*(w>>1) .. +32 (2 m-frags), cols 64*(w&1) .. +64 (4 n-frags)
// -> 8 WMMAs per wave per K-step.  M,N % 128 == 0, K % 32 == 0.
// ---------------------------------------------------------------------------
template <typename TIN, typename TOUT>
__global__ __launch_bounds__(256) void gemm_bias_wmma(
    const TIN* __restrict__ A, const float* __restrict__ W,
    const float* __restrict__ bias, TOUT* __restrict__ C,
    int M, int K, int N) {
  __shared__ alignas(16) bf16_t lA[2][128 * 32];  // A tile, row-major [m][k]
  __shared__ alignas(16) bf16_t lB[2][128 * 32];  // W^T tile, row-major [n][k]

  const int tid  = threadIdx.x;
  const int lane = tid & 31;
  const int w    = tid >> 5;
  const int wr   = w >> 1;      // 0..3 -> m offset 32*wr
  const int wc   = w & 1;       // 0..1 -> n offset 64*wc
  const int hh   = lane >> 4;   // lane half (0/1)
  const int l16  = lane & 15;
  const int m0   = blockIdx.y * 128;
  const int n0   = blockIdx.x * 128;

  // Register prefetch buffers (vec4 granularity; 4096 elems per tile,
  // 1024 vec4 across 256 threads -> 4 per thread).
  V4<TIN>   ra[4];
  V4<float> rb[4];

  auto load_tiles = [&](int k0) {
#pragma unroll
    for (int i = 0; i < 4; ++i) {
      int ia = tid + i * 256;
      int r = ia >> 3, c4 = ia & 7;       // A: 8 vec4 per 32-wide row
      ra[i] = *(const V4<TIN>*)&A[(size_t)(m0 + r) * K + k0 + 4 * c4];
      int kk = ia >> 5, n4 = ia & 31;     // W: 32 vec4 per 128-wide row
      rb[i] = *(const V4<float>*)&W[(size_t)(k0 + kk) * N + n0 + 4 * n4];
    }
  };
  auto store_tiles = [&](int buf) {
#pragma unroll
    for (int i = 0; i < 4; ++i) {
      int ia = tid + i * 256;
      int r = ia >> 3, c4 = ia & 7;
      bf16x4 t;
#pragma unroll
      for (int j = 0; j < 4; ++j) t.v[j] = cvt_bf16(ra[i].v[j]);
      *(bf16x4*)&lA[buf][r * 32 + 4 * c4] = t;
      int kk = ia >> 5, n4 = ia & 31;
#pragma unroll
      for (int j = 0; j < 4; ++j)
        lB[buf][(4 * n4 + j) * 32 + kk] = cvt_bf16(rb[i].v[j]);
    }
  };

  load_tiles(0);
  store_tiles(0);
  int cur = 0;

  v8f acc[2][4] = {};

  for (int k0 = 0; k0 < K; k0 += 32) {
    __syncthreads();  // tile `cur` visible; previous reads of `cur^1` done
    const bool more = (k0 + 32) < K;
    if (more) load_tiles(k0 + 32);  // global loads overlap the WMMAs below

    v16bf af[2], bfr[4];
#pragma unroll
    for (int mi = 0; mi < 2; ++mi) {
      const bf16_t* row = &lA[cur][(32 * wr + 16 * mi + l16) * 32];
      af[mi] = cat16(*(const v8bf*)(row + 8 * hh),          // k = 8h..
                     *(const v8bf*)(row + 16 + 8 * hh));    // k = 16+8h..
    }
#pragma unroll
    for (int ni = 0; ni < 4; ++ni) {
      const bf16_t* row = &lB[cur][(64 * wc + 16 * ni + l16) * 32];
      bfr[ni] = cat16(*(const v8bf*)(row + 8 * hh),         // k = 8h..
                      *(const v8bf*)(row + 16 + 8 * hh));
    }
#pragma unroll
    for (int mi = 0; mi < 2; ++mi)
#pragma unroll
      for (int ni = 0; ni < 4; ++ni)
        acc[mi][ni] = wmma_bf16(af[mi], bfr[ni], acc[mi][ni]);

    if (more) store_tiles(cur ^ 1);  // write other buffer: no barrier needed
    cur ^= 1;
  }

  // Epilogue: bias + store.  C elem r <-> row r + 8*half, col = lane%16.
#pragma unroll
  for (int mi = 0; mi < 2; ++mi)
#pragma unroll
    for (int ni = 0; ni < 4; ++ni) {
      int n = n0 + 64 * wc + 16 * ni + l16;
      float bv = bias[n];
#pragma unroll
      for (int r = 0; r < 8; ++r) {
        int m = m0 + 32 * wr + 16 * mi + r + 8 * hh;
        store_out(&C[(size_t)m * N + n], acc[mi][ni][r] + bv);
      }
    }
}

// ---------------------------------------------------------------------------
// Flash attention per (b, h): out = softmax(scale * Q K^T) V
// Q: (512,128) rows b*512+l of qb; K,V: (2048,128) rows of kb/vb (col h*128+e).
// Block = 128 threads (4 waves), 64 L-rows per block (16 per wave).
// S processed in 32-wide tiles; online softmax; O kept in 8 v8f accumulators.
// ---------------------------------------------------------------------------
__global__ __launch_bounds__(128) void attn_fmha(
    const bf16_t* __restrict__ qb, const bf16_t* __restrict__ kb,
    const bf16_t* __restrict__ vb, bf16_t* __restrict__ attn) {
  __shared__ alignas(16) bf16_t lQ[64 * 128];    // [l][e]
  __shared__ alignas(16) bf16_t lK[32 * 128];    // [s][e]  (= B^T for Q K^T)
  __shared__ alignas(16) bf16_t lVt[128 * 32];   // [e][s]  (= B^T for P V)
  __shared__ alignas(16) bf16_t lP[4][16 * 32];  // per-wave P, [m][s]

  const int tid  = threadIdx.x;
  const int lane = tid & 31;
  const int w    = tid >> 5;
  const int hh   = lane >> 4;
  const int l16  = lane & 15;
  const int b    = blockIdx.y >> 3;
  const int h    = blockIdx.y & 7;
  const int l0   = blockIdx.x * 64;
  const float scale = 0.08838834764831845f;  // 1/sqrt(128)

  // Stage Q tile (64x128), 8192 elems, 64 per thread.
#pragma unroll 8
  for (int i = 0; i < 64; ++i) {
    int idx = tid + i * 128;
    int r = idx >> 7, e = idx & 127;
    lQ[idx] = qb[(size_t)(b * 512 + l0 + r) * 1024 + h * 128 + e];
  }

  v8f o[8] = {};
  float mrun[8], lrun[8];
#pragma unroll
  for (int r = 0; r < 8; ++r) { mrun[r] = -INFINITY; lrun[r] = 0.0f; }

  for (int s0 = 0; s0 < 2048; s0 += 32) {
    __syncthreads();  // previous P@V done before overwriting lK/lVt
    // Stage K (32x128) and V^T (128x32); 4096 elems each, 32 per thread.
#pragma unroll 8
    for (int i = 0; i < 32; ++i) {
      int idx = tid + i * 128;
      int s = idx >> 7, e = idx & 127;
      size_t g = (size_t)(s0 + s) * 1024 + h * 128 + e;
      lK[idx]          = kb[g];
      lVt[e * 32 + s]  = vb[g];
    }
    __syncthreads();

    // scores: 16 rows (this wave) x 32 cols, K-dim = E = 128 (4 steps)
    v8f sc[2] = {};
#pragma unroll
    for (int ks = 0; ks < 4; ++ks) {
      const bf16_t* arow = &lQ[(16 * w + l16) * 128 + 32 * ks];
      v16bf afr = cat16(*(const v8bf*)(arow + 8 * hh),
                        *(const v8bf*)(arow + 16 + 8 * hh));
#pragma unroll
      for (int ni = 0; ni < 2; ++ni) {
        const bf16_t* brow = &lK[(16 * ni + l16) * 128 + 32 * ks + 8 * hh];
        v16bf bfr = cat16(*(const v8bf*)(brow), *(const v8bf*)(brow + 16));
        sc[ni] = wmma_bf16(afr, bfr, sc[ni]);
      }
    }

    // Online softmax update. Row r+8*hh lives in elem r across a 16-lane group,
    // so xor-reductions with masks 1..8 stay within the correct group.
#pragma unroll
    for (int r = 0; r < 8; ++r) {
      float s0v = sc[0][r] * scale;
      float s1v = sc[1][r] * scale;
      float mx = fmaxf(s0v, s1v);
#pragma unroll
      for (int d = 1; d < 16; d <<= 1) mx = fmaxf(mx, __shfl_xor(mx, d, 32));
      float mnew  = fmaxf(mrun[r], mx);
      float alpha = __expf(mrun[r] - mnew);
      float p0 = __expf(s0v - mnew);
      float p1 = __expf(s1v - mnew);
      float rs = p0 + p1;
#pragma unroll
      for (int d = 1; d < 16; d <<= 1) rs += __shfl_xor(rs, d, 32);
      lrun[r] = lrun[r] * alpha + rs;
      mrun[r] = mnew;
#pragma unroll
      for (int nf = 0; nf < 8; ++nf) o[nf][r] *= alpha;
      // C-layout -> A-layout via LDS: row m = r+8*hh, cols l16 and 16+l16
      int mrow = r + 8 * hh;
      lP[w][mrow * 32 + l16]      = (bf16_t)p0;
      lP[w][mrow * 32 + 16 + l16] = (bf16_t)p1;
    }
    __syncthreads();  // lP visible (block-wide sync also orders within wave)

    // P@V: A = lP[w] (16x32), B^T = lVt, N = 128 -> 8 fragments, one K-step.
    {
      const bf16_t* arow = &lP[w][l16 * 32];
      v16bf afr = cat16(*(const v8bf*)(arow + 8 * hh),
                        *(const v8bf*)(arow + 16 + 8 * hh));
#pragma unroll
      for (int nf = 0; nf < 8; ++nf) {
        const bf16_t* brow = &lVt[(16 * nf + l16) * 32 + 8 * hh];
        v16bf bfr = cat16(*(const v8bf*)(brow), *(const v8bf*)(brow + 16));
        o[nf] = wmma_bf16(afr, bfr, o[nf]);
      }
    }
  }

  // Normalize and store attn (bf16) in (B*L, H*E) layout for the O-projection.
#pragma unroll
  for (int nf = 0; nf < 8; ++nf) {
    int e = 16 * nf + l16;
#pragma unroll
    for (int r = 0; r < 8; ++r) {
      int lrow = l0 + 16 * w + r + 8 * hh;
      float val = o[nf][r] / lrun[r];
      attn[(size_t)(b * 512 + lrow) * 1024 + h * 128 + e] = (bf16_t)val;
    }
  }
}

// ---------------------------------------------------------------------------
// Host-side orchestration.
//   B=8, L=512, S=2048, D_MODEL=1024, D_LLM=4096, H=8, E=128
// Workspace layout (bytes):
//   qb   @ 0        : 4096*1024 bf16 = 8 MB
//   kb   @ 8  MB    : 2048*1024 bf16 = 4 MB
//   vb   @ 12 MB    : 2048*1024 bf16 = 4 MB
//   attn @ 16 MB    : 4096*1024 bf16 = 8 MB   (24 MB total)
// ---------------------------------------------------------------------------
extern "C" void kernel_launch(void* const* d_in, const int* in_sizes, int n_in,
                              void* d_out, int out_size, void* d_ws,
                              size_t ws_size, hipStream_t stream) {
  (void)in_sizes; (void)n_in; (void)out_size; (void)ws_size;
  const float* target = (const float*)d_in[0];
  const float* source = (const float*)d_in[1];
  const float* value  = (const float*)d_in[2];
  const float* Wq = (const float*)d_in[3];
  const float* bq = (const float*)d_in[4];
  const float* Wk = (const float*)d_in[5];
  const float* bk = (const float*)d_in[6];
  const float* Wv = (const float*)d_in[7];
  const float* bv = (const float*)d_in[8];
  const float* Wo = (const float*)d_in[9];
  const float* bo = (const float*)d_in[10];
  float* out = (float*)d_out;

  char* ws = (char*)d_ws;
  bf16_t* qb   = (bf16_t*)(ws);
  bf16_t* kb   = (bf16_t*)(ws + (size_t)8  * 1024 * 1024);
  bf16_t* vb   = (bf16_t*)(ws + (size_t)12 * 1024 * 1024);
  bf16_t* attn = (bf16_t*)(ws + (size_t)16 * 1024 * 1024);

  dim3 blk(256);
  // Q = target @ Wq + bq : (4096,1024)x(1024,1024)
  gemm_bias_wmma<float, bf16_t>
      <<<dim3(1024 / 128, 4096 / 128), blk, 0, stream>>>(target, Wq, bq, qb,
                                                         4096, 1024, 1024);
  // K = source @ Wk + bk : (2048,4096)x(4096,1024)
  gemm_bias_wmma<float, bf16_t>
      <<<dim3(1024 / 128, 2048 / 128), blk, 0, stream>>>(source, Wk, bk, kb,
                                                         2048, 4096, 1024);
  // V = value @ Wv + bv
  gemm_bias_wmma<float, bf16_t>
      <<<dim3(1024 / 128, 2048 / 128), blk, 0, stream>>>(value, Wv, bv, vb,
                                                         2048, 4096, 1024);
  // Attention: grid = (L/64, B*H)
  attn_fmha<<<dim3(512 / 64, 8 * 8), dim3(128), 0, stream>>>(qb, kb, vb, attn);
  // Out = attn @ Wo + bo : (4096,1024)x(1024,4096), f32 output
  gemm_bias_wmma<bf16_t, float>
      <<<dim3(4096 / 128, 4096 / 128), blk, 0, stream>>>(attn, Wo, bo, out,
                                                         4096, 1024, 4096);
}
